// TemporalAxialAttention_67070209294657
// MI455X (gfx1250) — matrix operand order repaired and checked by reading, not compiled, not run on hardware
//
#include <hip/hip_runtime.h>

// ---------------------------------------------------------------------------
// Types for CDNA5 WMMA (wave32): v_wmma_f32_16x16x32_bf16
// ---------------------------------------------------------------------------
typedef __bf16 bf16x16 __attribute__((ext_vector_type(16)));
typedef float  floatx8 __attribute__((ext_vector_type(8)));
typedef unsigned short ushort8 __attribute__((ext_vector_type(8)));

union FragU { bf16x16 v; ushort8 h[2]; };

__device__ __forceinline__ unsigned short f2bf(float f) {
  unsigned int u = __float_as_uint(f);
  u += 0x7fffu + ((u >> 16) & 1u);          // round-to-nearest-even
  return (unsigned short)(u >> 16);
}

__device__ __forceinline__ float sigma_act(float x) {   // elu(x)+1
  return x > 0.f ? x + 1.f : __expf(x);
}

// A-matrix 16x32 bf16 fragment (ISA 7.12.2): lane r=lane&15, h=lane>>4.
// elements 0..7  = K[8h .. 8h+7], elements 8..15 = K[16+8h .. 23+8h].
// `base` points at tile origin of a K-contiguous (row-major) bf16 buffer.
__device__ __forceinline__ bf16x16 load_a_frag(const unsigned short* base, int stride, int lane) {
  const int r = lane & 15, h = lane >> 4;
  FragU f;
  f.h[0] = *(const ushort8*)(base + r * stride + 8 * h);
  f.h[1] = *(const ushort8*)(base + r * stride + 16 + 8 * h);
  return f.v;
}

// B-matrix 32x16 bf16 fragment: lane c=lane&15 selects column N=c; h=lane>>4:
// elements 0..7 = K[8h .. 8h+7], elements 8..15 = K[16+8h .. 23+8h].
// `base` points at tile origin of BT, an N-major / K-contiguous bf16 buffer.
__device__ __forceinline__ bf16x16 load_b_frag(const unsigned short* base, int stride, int lane) {
  const int c = lane & 15, h = lane >> 4;
  FragU f;
  f.h[0] = *(const ushort8*)(base + c * stride + 8 * h);
  f.h[1] = *(const ushort8*)(base + c * stride + 16 + 8 * h);
  return f.v;
}

__device__ __forceinline__ floatx8 wmma_bf16(bf16x16 a, bf16x16 b, floatx8 c) {
  return __builtin_amdgcn_wmma_f32_16x16x32_bf16(false, a, false, b, (short)0, c, false, false);
}

// ---------------------------------------------------------------------------
// Problem constants
// ---------------------------------------------------------------------------
#define BB 2
#define TT 32
#define HH_SP 24
#define WW_SP 24
#define DD 1024
#define HEADS 16
#define DHEAD 64
#define STRIDE_S 8
#define M_ROWS (BB*TT*HH_SP*WW_SP)        // 36864
#define A_BATCH (BB*HH_SP*WW_SP)          // 1152 axial batches
#define T_STRIDE (HH_SP*WW_SP)            // 576 row stride per t

// ---------------------------------------------------------------------------
// Conversion kernels
// ---------------------------------------------------------------------------
__global__ __launch_bounds__(256) void k_f32_to_bf16(
    const float* __restrict__ src, unsigned short* __restrict__ dst, int n) {
  int i = blockIdx.x * 256 + threadIdx.x;
  if (i < n) dst[i] = f2bf(src[i]);
}

// W (K x N row-major, f32) -> WT (N x K, bf16) so GEMM B fragments are K-contiguous
__global__ __launch_bounds__(256) void k_transpose_bf16(
    const float* __restrict__ W, unsigned short* __restrict__ WT, int K, int N) {
  int i = blockIdx.x * 256 + threadIdx.x;
  if (i < K * N) {
    int k = i / N, n = i - k * N;
    WT[(size_t)n * K + k] = f2bf(W[i]);
  }
}

// ---------------------------------------------------------------------------
// Tiled WMMA GEMM: C[M,N] = A[M,K]_bf16 * BT[N,K]_bf16^T (+ bias[n])
// block = 256 threads (8 waves), tile 128x128, K-step 32.
// Register-staged DOUBLE BUFFERING: next-slab global loads are issued before
// the wmma burst so the 8 wmmas hide global-load latency; LDS stores to the
// alternate buffer happen after compute, one barrier per iteration.
// wave (w&3) -> 32-row band, (w>>2) -> 64-col band; 2x4 accum tiles of 16x16.
// ---------------------------------------------------------------------------
__global__ __launch_bounds__(256) void k_gemm_bf16(
    const unsigned short* __restrict__ A, const unsigned short* __restrict__ BT,
    const float* __restrict__ bias, float* __restrict__ C,
    int M, int N, int K) {
  __shared__ __align__(16) unsigned short Asm[2][128][40];
  __shared__ __align__(16) unsigned short Bsm[2][128][40];

  const int tid  = threadIdx.x;
  const int lane = tid & 31;
  const int w    = tid >> 5;
  const int wm   = w & 3;       // row band
  const int wn   = w >> 2;      // col band
  const int m0   = blockIdx.y * 128;
  const int n0   = blockIdx.x * 128;

  // global->LDS staging map: thread covers rows rr and rr+64, one 16B segment
  const int rr  = tid >> 2;
  const int seg = (tid & 3) * 8;
  const size_t aoff0 = (size_t)(m0 + rr) * K + seg;
  const size_t aoff1 = (size_t)(m0 + rr + 64) * K + seg;
  const size_t boff0 = (size_t)(n0 + rr) * K + seg;
  const size_t boff1 = (size_t)(n0 + rr + 64) * K + seg;

  const floatx8 zf = {0.f,0.f,0.f,0.f,0.f,0.f,0.f,0.f};
  floatx8 acc[2][4];
  for (int i = 0; i < 2; ++i)
    for (int j = 0; j < 4; ++j) acc[i][j] = zf;

  // prologue: fill buffer 0
  {
    ushort8 ra0 = *(const ushort8*)&A[aoff0];
    ushort8 ra1 = *(const ushort8*)&A[aoff1];
    ushort8 rb0 = *(const ushort8*)&BT[boff0];
    ushort8 rb1 = *(const ushort8*)&BT[boff1];
    *(ushort8*)&Asm[0][rr][seg]      = ra0;
    *(ushort8*)&Asm[0][rr + 64][seg] = ra1;
    *(ushort8*)&Bsm[0][rr][seg]      = rb0;
    *(ushort8*)&Bsm[0][rr + 64][seg] = rb1;
  }
  __syncthreads();

  int cur = 0;
  for (int k0 = 0; k0 < K; k0 += 32) {
    const bool has_next = (k0 + 32 < K);
    ushort8 ra0, ra1, rb0, rb1;
    if (has_next) {   // issue next-slab loads BEFORE compute (latency hiding)
      const size_t ko = (size_t)(k0 + 32);
      ra0 = *(const ushort8*)&A[aoff0 + ko];
      ra1 = *(const ushort8*)&A[aoff1 + ko];
      rb0 = *(const ushort8*)&BT[boff0 + ko];
      rb1 = *(const ushort8*)&BT[boff1 + ko];
    }

    bf16x16 af[2], bf_[4];
    for (int mt = 0; mt < 2; ++mt)
      af[mt] = load_a_frag(&Asm[cur][32 * wm + 16 * mt][0], 40, lane);
    for (int nt = 0; nt < 4; ++nt)
      bf_[nt] = load_b_frag(&Bsm[cur][64 * wn + 16 * nt][0], 40, lane);
    for (int mt = 0; mt < 2; ++mt)
      for (int nt = 0; nt < 4; ++nt)
        acc[mt][nt] = wmma_bf16(af[mt], bf_[nt], acc[mt][nt]);

    if (has_next) {   // write the other buffer; prev readers passed last barrier
      const int nb = cur ^ 1;
      *(ushort8*)&Asm[nb][rr][seg]      = ra0;
      *(ushort8*)&Asm[nb][rr + 64][seg] = ra1;
      *(ushort8*)&Bsm[nb][rr][seg]      = rb0;
      *(ushort8*)&Bsm[nb][rr + 64][seg] = rb1;
      __syncthreads();
      cur = nb;
    }
  }

  // C/D layout: lane c=lane&15 -> N col; VGPR v -> M row v + 8*(lane>>4)
  const int c = lane & 15, hh = lane >> 4;
  for (int mt = 0; mt < 2; ++mt)
    for (int nt = 0; nt < 4; ++nt)
      for (int v = 0; v < 8; ++v) {
        int mrow = m0 + 32 * wm + 16 * mt + v + 8 * hh;
        int col  = n0 + 64 * wn + 16 * nt + c;
        float val = acc[mt][nt][v];
        if (bias) val += bias[col];
        C[(size_t)mrow * N + col] = val;
      }
}

// ---------------------------------------------------------------------------
// Fused attention per (axial-batch, head): 18432 blocks x 256 threads.
// ---------------------------------------------------------------------------
__global__ __launch_bounds__(256) void k_attention(
    const float* __restrict__ qkv,       // [36864, 3072]
    const float* __restrict__ mem,       // [1152,16,64,64]
    const float* __restrict__ z,         // [1152,16,64]
    const float* __restrict__ mem_beta,  // [16]
    const float* __restrict__ inv_freq,  // [32]
    unsigned short* __restrict__ out_bf, // [36864,1024] bf16 activations
    float* __restrict__ new_mem, float* __restrict__ new_z) {

  __shared__ float q_lds[32][64];
  __shared__ float k_lds[32][64];
  __shared__ float v_lds[32][64];
  __shared__ float sig_ks[8][64];
  __shared__ float scores_lds[32][33];
  __shared__ __align__(16) unsigned short q_bf[32][72];
  __shared__ __align__(16) unsigned short k_bf[32][72];
  __shared__ __align__(16) unsigned short sq_bf[32][72];
  __shared__ __align__(16) unsigned short attn_bf[32][40];
  __shared__ __align__(16) unsigned short vT_bf[64][40];
  __shared__ __align__(16) unsigned short memT_bf[64][72];
  __shared__ float denom_lds[32];

  const int bh   = blockIdx.x;
  const int a    = bh >> 4;
  const int head = bh & 15;
  const int b    = a / (HH_SP * WW_SP);
  const int rem  = a - b * (HH_SP * WW_SP);
  const int mbase = ((b * TT) * HH_SP + (rem / WW_SP)) * WW_SP + (rem % WW_SP);
  const int tid = threadIdx.x, lane = tid & 31, w = tid >> 5;
  const int colq = head * DHEAD, colk = 1024 + head * DHEAD, colv = 2048 + head * DHEAD;

  // --- load q,k,v tiles (32x64 f32 each), coalesced over d ---
  for (int i = tid; i < 2048; i += 256) {
    int t = i >> 6, d = i & 63;
    size_t ro = (size_t)(mbase + t * T_STRIDE) * 3072;
    q_lds[t][d] = qkv[ro + colq + d];
    k_lds[t][d] = qkv[ro + colk + d];
    v_lds[t][d] = qkv[ro + colv + d];
  }
  __syncthreads();

  // --- new_z = z + sum_t sigma(k)  (pre-rotary) ---
  const size_t zbase = (size_t)bh * 64;
  if (tid < 64) {
    float accz = z[zbase + tid];
    for (int t = 0; t < TT; ++t) accz += sigma_act(k_lds[t][tid]);
    new_z[zbase + tid] = accz;
  }
  // reference applies sigma twice on the memory-update keys:
  // new_mem = mem + sigma(sigma(ks)^T) @ vs
  for (int i = tid; i < STRIDE_S * 64; i += 256) {
    int s = i >> 6, d = i & 63;
    sig_ks[s][d] = sigma_act(sigma_act(k_lds[s][d]));
  }
  __syncthreads();

  // --- new_mem + stage old mem transposed as bf16 for retrieval WMMA ---
  const size_t membase = (size_t)bh * 4096;
  for (int i = tid; i < 4096; i += 256) {
    int kk = i >> 6, vv = i & 63;
    float mv = mem[membase + i];
    memT_bf[vv][kk] = f2bf(mv);           // BT layout for sigma_q @ mem
    float accm = mv;
    for (int s = 0; s < STRIDE_S; ++s) accm += sig_ks[s][kk] * v_lds[s][vv];
    new_mem[membase + i] = accm;
  }
  __syncthreads();

  // --- rotary embedding in place on q,k ---
  for (int i = tid; i < 1024; i += 256) {
    int t = i >> 5, j = i & 31;
    float f = (float)t * inv_freq[j];
    float cf = cosf(f), sf = sinf(f);
    float x1 = q_lds[t][2 * j], x2 = q_lds[t][2 * j + 1];
    q_lds[t][2 * j]     = x1 * cf - x2 * sf;
    q_lds[t][2 * j + 1] = x2 * cf + x1 * sf;
    x1 = k_lds[t][2 * j]; x2 = k_lds[t][2 * j + 1];
    k_lds[t][2 * j]     = x1 * cf - x2 * sf;
    k_lds[t][2 * j + 1] = x2 * cf + x1 * sf;
  }
  __syncthreads();

  // --- bf16 conversions + denom = sigma(q_rot) . z ---
  for (int i = tid; i < 2048; i += 256) {
    int t = i >> 6, d = i & 63;
    float qv = q_lds[t][d];
    q_bf[t][d]  = f2bf(qv);
    k_bf[t][d]  = f2bf(k_lds[t][d]);
    sq_bf[t][d] = f2bf(sigma_act(qv));
    vT_bf[d][t] = f2bf(v_lds[t][d]);      // v stays un-rotated
  }
  if (tid < 32) {
    float accd = 0.f;
    for (int d = 0; d < 64; ++d) accd += sigma_act(q_lds[tid][d]) * z[zbase + d];
    denom_lds[tid] = accd;
  }
  __syncthreads();

  // --- scores = q_rot @ k_rot^T : waves 0..3 each one 16x16 tile, K=64 ---
  if (w < 4) {
    int ti = w >> 1, tj = w & 1;
    const floatx8 zf = {0.f,0.f,0.f,0.f,0.f,0.f,0.f,0.f};
    floatx8 accs = zf;
    for (int kb = 0; kb < 64; kb += 32) {
      bf16x16 afr = load_a_frag(&q_bf[16 * ti][kb], 72, lane);
      bf16x16 bfr = load_b_frag(&k_bf[16 * tj][kb], 72, lane);
      accs = wmma_bf16(afr, bfr, accs);
    }
    int c = lane & 15, h2 = lane >> 4;
    for (int v = 0; v < 8; ++v)
      scores_lds[16 * ti + v + 8 * h2][16 * tj + c] = accs[v];
  }
  __syncthreads();

  // --- causal softmax rows (scale = 1/sqrt(64)) ---
  if (tid < 32) {
    int t = tid;
    float mx = -1e30f;
    for (int s = 0; s <= t; ++s) mx = fmaxf(mx, scores_lds[t][s] * 0.125f);
    float sum = 0.f;
    for (int s = 0; s <= t; ++s) sum += __expf(scores_lds[t][s] * 0.125f - mx);
    float inv = 1.f / sum;
    for (int s = 0; s < 32; ++s)
      attn_bf[t][s] = (s <= t) ? f2bf(__expf(scores_lds[t][s] * 0.125f - mx) * inv)
                               : (unsigned short)0;
  }
  __syncthreads();

  // --- each wave: one 16x16 (t,d) output tile; fuse attn@v and sigma_q@mem ---
  {
    int ti = w >> 2, tj = w & 3;           // t-tile (0..1), d-tile (0..3)
    const floatx8 zf = {0.f,0.f,0.f,0.f,0.f,0.f,0.f,0.f};
    floatx8 accO = zf;                     // attn @ v, K=32
    {
      bf16x16 aA = load_a_frag(&attn_bf[16 * ti][0], 40, lane);
      bf16x16 bV = load_b_frag(&vT_bf[16 * tj][0], 40, lane);
      accO = wmma_bf16(aA, bV, accO);
    }
    floatx8 accR = zf;                     // sigma(q) @ mem, K=64
    for (int kb = 0; kb < 64; kb += 32) {
      bf16x16 aQ = load_a_frag(&sq_bf[16 * ti][kb], 72, lane);
      bf16x16 bM = load_b_frag(&memT_bf[16 * tj][kb], 72, lane);
      accR = wmma_bf16(aQ, bM, accR);
    }
    float g = 1.f / (1.f + __expf(-mem_beta[head]));
    int c = lane & 15, h2 = lane >> 4;
    for (int v = 0; v < 8; ++v) {
      int t = 16 * ti + v + 8 * h2;
      int d = 16 * tj + c;
      float mem_x = accR[v] / denom_lds[t];
      float o = g * mem_x + (1.f - g) * accO[v];
      out_bf[(size_t)(mbase + t * T_STRIDE) * 1024 + head * DHEAD + d] = f2bf(o);
    }
  }
}

// ---------------------------------------------------------------------------
// Launch
// ---------------------------------------------------------------------------
extern "C" void kernel_launch(void* const* d_in, const int* in_sizes, int n_in,
                              void* d_out, int out_size, void* d_ws, size_t ws_size,
                              hipStream_t stream) {
  const float* x        = (const float*)d_in[0];
  const float* mem      = (const float*)d_in[1];
  const float* z        = (const float*)d_in[2];
  const float* W_qkv    = (const float*)d_in[3];
  const float* W_out    = (const float*)d_in[4];
  const float* b_out    = (const float*)d_in[5];
  const float* mem_beta = (const float*)d_in[6];
  const float* inv_freq = (const float*)d_in[7];

  float* outp   = (float*)d_out;
  float* x_out  = outp;                                   // 36864*1024
  float* nmem   = outp + (size_t)M_ROWS * DD;             // 1152*16*64*64
  float* nz     = nmem + (size_t)A_BATCH * HEADS * DHEAD * DHEAD;

  char* ws = (char*)d_ws;
  unsigned short* x_bf   = (unsigned short*)(ws);                       //  75,497,472 B
  unsigned short* wqkvT  = (unsigned short*)(ws + 75497472u);           //   6,291,456 B
  unsigned short* woutT  = (unsigned short*)(ws + 81788928u);           //   2,097,152 B
  float*          qkv    = (float*)         (ws + 83886080u);           // 452,984,832 B
  unsigned short* act_bf = (unsigned short*)(ws + 536870912u);          //  75,497,472 B

  // 1) x -> bf16 (row-major M x 1024)
  {
    int n = M_ROWS * DD;
    k_f32_to_bf16<<<(n + 255) / 256, 256, 0, stream>>>(x, x_bf, n);
  }
  // 2) weights -> transposed bf16 (N-major, K-contiguous)
  {
    int n1 = DD * 3 * DD;
    k_transpose_bf16<<<(n1 + 255) / 256, 256, 0, stream>>>(W_qkv, wqkvT, DD, 3 * DD);
    int n2 = DD * DD;
    k_transpose_bf16<<<(n2 + 255) / 256, 256, 0, stream>>>(W_out, woutT, DD, DD);
  }
  // 3) QKV projection GEMM: [36864,1024] x [1024,3072]
  {
    dim3 grid(3 * DD / 128, M_ROWS / 128);
    k_gemm_bf16<<<grid, 256, 0, stream>>>(x_bf, wqkvT, nullptr, qkv,
                                          M_ROWS, 3 * DD, DD);
  }
  // 4) fused attention + linear-memory per (axial batch, head)
  {
    k_attention<<<A_BATCH * HEADS, 256, 0, stream>>>(
        qkv, mem, z, mem_beta, inv_freq, act_bf, nmem, nz);
  }
  // 5) output projection GEMM (+bias) into d_out
  {
    dim3 grid(DD / 128, M_ROWS / 128);
    k_gemm_bf16<<<grid, 256, 0, stream>>>(act_bf, woutT, b_out, x_out,
                                          M_ROWS, DD, DD);
  }
}
